// MultiScaleConv_63556926046792
// MI455X (gfx1250) — matrix operand order repaired and checked by reading, not compiled, not run on hardware
//
#include <hip/hip_runtime.h>

typedef _Float16 v16h __attribute__((ext_vector_type(16)));
typedef _Float16 v2h  __attribute__((ext_vector_type(2)));
typedef float    v8f  __attribute__((ext_vector_type(8)));
typedef int      v4i  __attribute__((ext_vector_type(4)));
typedef int      v2i  __attribute__((ext_vector_type(2)));

#define LSEQ 8192
#define CCH  64
#define TILE 64
#define NT   4    // L-tiles per block (weight-swizzle amortization + async double buffering)
#define XS   72   // input tile stride (floats): 4 pad | 64 interior | 4 pad; row = 288 B (16B-mult)
#define YS   80   // f16 LDS stride: row = 160 B, multiple of 32 -> aligned v16h fragments

#if defined(__has_builtin)
#  if __has_builtin(__builtin_amdgcn_global_load_async_to_lds_b128) && \
      __has_builtin(__builtin_amdgcn_global_load_async_to_lds_b64) &&  \
      __has_builtin(__builtin_amdgcn_s_wait_asynccnt)
#    define USE_ASYNC_LDS 1
#  endif
#endif

#ifdef USE_ASYNC_LDS
typedef __attribute__((address_space(1))) v4i gv4i;
typedef __attribute__((address_space(3))) v4i lv4i;
typedef __attribute__((address_space(1))) v2i gv2i;
typedef __attribute__((address_space(3))) v2i lv2i;

__device__ __forceinline__ gv4i* glb4(const void* p) { return (gv4i*)(uintptr_t)p; }
__device__ __forceinline__ gv2i* glb2(const void* p) { return (gv2i*)(uintptr_t)p; }
// generic LDS pointer: low 32 bits are the LDS byte offset (flat aperture rule)
__device__ __forceinline__ lv4i* lds4(void* p) { return (lv4i*)(uint32_t)(uintptr_t)p; }
__device__ __forceinline__ lv2i* lds2(void* p) { return (lv2i*)(uint32_t)(uintptr_t)p; }

// Stage one 64ch x (64+halo) tile into `buf`. Every wave issues EXACTLY 5 async
// instructions (4 interior b128 + 1 halo b64), so ASYNCcnt-based partial waits
// are well defined. OOB halo lanes write zeros via DS instead (disjoint addrs).
__device__ __forceinline__ void stage_async(const float* __restrict__ x, int bi,
                                            int l0, float* buf, int tid) {
    #pragma unroll
    for (int i = 0; i < 4; ++i) {              // 4*256 = 64 rows * 16 chunks
        int idx = i * 256 + tid;
        int row = idx >> 4;
        int c4  = idx & 15;
        __builtin_amdgcn_global_load_async_to_lds_b128(
            glb4(&x[(bi * CCH + row) * LSEQ + l0 + c4 * 4]),
            lds4(&buf[row * XS + 4 + c4 * 4]), 0, 0);
    }
    // 128 halo tasks on even lanes -> 16 active lanes in every wave
    int task = tid >> 1;
    int row  = task >> 1;
    int side = task & 1;
    int p    = side ? (l0 + TILE) : (l0 - 2);
    int dcol = side ? (4 + TILE) : 2;
    bool own   = (tid & 1) == 0;
    bool valid = (p >= 0) && (p + 1 < LSEQ);
    if (own && valid) {
        __builtin_amdgcn_global_load_async_to_lds_b64(
            glb2(&x[(bi * CCH + row) * LSEQ + p]),
            lds2(&buf[row * XS + dcol]), 0, 0);
    }
    if (own && !valid) {
        buf[row * XS + dcol]     = 0.0f;
        buf[row * XS + dcol + 1] = 0.0f;
    }
}
#else
__device__ __forceinline__ void stage_sync(const float* __restrict__ x, int bi,
                                           int l0, float* buf, int tid) {
    #pragma unroll
    for (int i = 0; i < 17; ++i) {             // 17*256 = 64*68
        int idx = i * 256 + tid;
        int row = idx / 68;
        int col = idx - row * 68;
        int p   = l0 + col - 2;
        float vv = 0.0f;
        if (p >= 0 && p < LSEQ) vv = x[(bi * CCH + row) * LSEQ + p];
        buf[row * XS + 2 + col] = vv;
    }
    if (l0 + TILE < LSEQ) {
        __builtin_prefetch(&x[(bi * CCH + (tid >> 2)) * LSEQ + l0 + TILE + ((tid & 3) << 4)], 0, 1);
    }
}
#endif

// Fused branch: dwconv(KW,PAD) -> [64x64 GEMM + bias + relu] -> [64x64 GEMM + bias] -> out
// One block = one batch x NT consecutive 64-position tiles. 256 threads = 8 waves (wave32).
template <int KW, int PAD>
__global__ __launch_bounds__(256) void msconv_branch(
    const float* __restrict__ x,      // [B, C, L]
    const float* __restrict__ dw_w,   // [C, 1, KW]
    const float* __restrict__ dw_b,   // [C]
    const float* __restrict__ g_w,    // [C, C]
    const float* __restrict__ g_b,    // [C]
    const float* __restrict__ lin_w,  // [C, C]
    const float* __restrict__ lin_b,  // [C]
    float* __restrict__ out)          // [B, C, L]
{
    __shared__ float    xs0[CCH * XS];    // input tile double buffer (f32)
    __shared__ float    xs1[CCH * XS];
    __shared__ _Float16 wg[CCH * YS];     // g_w, A-fragment swizzled
    __shared__ _Float16 wl[CCH * YS];     // lin_w, A-fragment swizzled
    __shared__ _Float16 ysw[TILE * YS];   // dwconv out, [pos][chan] (B-fragment order)
    __shared__ _Float16 zsw[TILE * YS];   // relu intermediate, [pos][chan]

    const int tid     = threadIdx.x;
    const int l0_base = blockIdx.x * (TILE * NT);
    const int bi      = blockIdx.y;

#ifdef USE_ASYNC_LDS
    stage_async(x, bi, l0_base, xs0, tid);     // kick off tile 0 immediately
#endif

    // ---- swizzle both weight matrices into A-fragment order, f32 -> f16 ----
    // (done ONCE per block, amortized over NT tiles)
    // A operand k order used here (contiguous, same as B; not AMD's table order), lane half h, k-step ks:
    //   element e (0..15) holds K = k0 + 16*h + e
    // store row m as 4 contiguous 16-half chunks: chunk = h + 2*ks
    #pragma unroll
    for (int i = 0; i < 16; ++i) {             // 16*256 = 64*64
        int idx = i * 256 + tid;
        int m = idx >> 6;
        int k = idx & 63;
        int ks = k >> 5;
        int r  = k & 31;
        int hh = r >> 4;
        int e  = r & 15;
        int dst = m * YS + (hh + 2 * ks) * 16 + e;
        wg[dst] = (_Float16)g_w[idx];
        wl[dst] = (_Float16)lin_w[idx];
    }

    // per-channel dwconv weights held in registers across all NT tiles
    const int cch   = tid >> 2;
    const int lbase = (tid & 3) << 4;
    float wdw[KW];
    #pragma unroll
    for (int j = 0; j < KW; ++j) wdw[j] = dw_w[cch * KW + j];
    const float bdw = dw_b[cch];

    const int lane = tid & 31;
    const int wv   = tid >> 5;
    const int lrow = lane & 15;
    const int h    = lane >> 4;

    for (int t = 0; t < NT; ++t) {
        float* cur = (t & 1) ? xs1 : xs0;
        const int l0 = l0_base + t * TILE;

#ifdef USE_ASYNC_LDS
        if (t + 1 < NT) {
            float* nxt = (t & 1) ? xs0 : xs1;
            stage_async(x, bi, l0 + TILE, nxt, tid);  // prefetch next tile
            __builtin_amdgcn_s_wait_asynccnt(5);      // drain current tile only
        } else {
            __builtin_amdgcn_s_wait_asynccnt(0);
        }
#else
        stage_sync(x, bi, l0, cur, tid);
#endif
        __syncthreads();

        // ---- depthwise conv on VALU; write f16 to ysw[pos][chan] ----
        #pragma unroll
        for (int i = 0; i < 16; ++i) {
            int l = lbase + i;
            float a = bdw;
            #pragma unroll
            for (int j = 0; j < KW; ++j)
                a += wdw[j] * cur[cch * XS + l + 4 - PAD + j];
            ysw[l * YS + cch] = (_Float16)a;
        }
        __syncthreads();

        // ---- GEMM1: z = relu(G * y + g_b), stays in LDS ----
        #pragma unroll
        for (int st = 0; st < 2; ++st) {
            int s  = wv + 8 * st;            // 16 subtiles over 8 waves
            int mT = (s >> 2) << 4;
            int nT = (s & 3) << 4;
            v8f acc = {};
            #pragma unroll
            for (int ks = 0; ks < 2; ++ks) {
                v16h a = *(const v16h*)&wg[(mT + lrow) * YS + (h + 2 * ks) * 16];
                v16h b = *(const v16h*)&ysw[(nT + lrow) * YS + ks * 32 + h * 16];
                acc = __builtin_amdgcn_wmma_f32_16x16x32_f16(
                    false, a, false, b, (short)0, acc, false, false);
            }
            int n     = nT + lrow;
            int mBase = mT + 8 * h;
            #pragma unroll
            for (int rp = 0; rp < 4; ++rp) {
                int M0 = mBase + 2 * rp;
                float v0 = fmaxf(acc[2 * rp]     + g_b[M0],     0.0f);
                float v1 = fmaxf(acc[2 * rp + 1] + g_b[M0 + 1], 0.0f);
                v2h pk = { (_Float16)v0, (_Float16)v1 };
                *(v2h*)&zsw[n * YS + M0] = pk;   // [pos][chan] = B-fragment order
            }
        }
        __syncthreads();

        // ---- GEMM2: out = W * z + lin_b, nontemporal f32 store to HBM ----
        #pragma unroll
        for (int st = 0; st < 2; ++st) {
            int s  = wv + 8 * st;
            int mT = (s >> 2) << 4;
            int nT = (s & 3) << 4;
            v8f acc = {};
            #pragma unroll
            for (int ks = 0; ks < 2; ++ks) {
                v16h a = *(const v16h*)&wl[(mT + lrow) * YS + (h + 2 * ks) * 16];
                v16h b = *(const v16h*)&zsw[(nT + lrow) * YS + ks * 32 + h * 16];
                acc = __builtin_amdgcn_wmma_f32_16x16x32_f16(
                    false, a, false, b, (short)0, acc, false, false);
            }
            int n     = nT + lrow;
            int mBase = mT + 8 * h;
            #pragma unroll
            for (int r = 0; r < 8; ++r) {
                int M = mBase + r;
                __builtin_nontemporal_store(acc[r] + lin_b[M],
                                            &out[(bi * CCH + M) * LSEQ + l0 + n]);
            }
        }
        __syncthreads();   // zsw/ysw reuse fence before next tile
    }
}

extern "C" void kernel_launch(void* const* d_in, const int* in_sizes, int n_in,
                              void* d_out, int out_size, void* d_ws, size_t ws_size,
                              hipStream_t stream) {
    (void)in_sizes; (void)n_in; (void)out_size; (void)d_ws; (void)ws_size;

    const float* k      = (const float*)d_in[0];
    const float* v      = (const float*)d_in[1];
    const float* dw3_w  = (const float*)d_in[2];
    const float* dw3_b  = (const float*)d_in[3];
    const float* g3_w   = (const float*)d_in[4];
    const float* g3_b   = (const float*)d_in[5];
    const float* lin3_w = (const float*)d_in[6];
    const float* lin3_b = (const float*)d_in[7];
    const float* dw5_w  = (const float*)d_in[8];
    const float* dw5_b  = (const float*)d_in[9];
    const float* g5_w   = (const float*)d_in[10];
    const float* g5_b   = (const float*)d_in[11];
    const float* lin5_w = (const float*)d_in[12];
    const float* lin5_b = (const float*)d_in[13];

    float* out = (float*)d_out;
    const size_t branch_elems = (size_t)64 * CCH * LSEQ;

    dim3 grid(LSEQ / (TILE * NT), 64);
    dim3 block(256);

    msconv_branch<3, 1><<<grid, block, 0, stream>>>(
        k, dw3_w, dw3_b, g3_w, g3_b, lin3_w, lin3_b, out);
    msconv_branch<5, 2><<<grid, block, 0, stream>>>(
        v, dw5_w, dw5_b, g5_w, g5_b, lin5_w, lin5_b, out + branch_elems);
}
